// GSNModel_86500641341859
// MI455X (gfx1250) — matrix operand order repaired and checked by reading, not compiled, not run on hardware
//
#include <hip/hip_runtime.h>
#include <hip/hip_bf16.h>
#include <math.h>

#define B_    16
#define S_    96
#define T_    50
#define E_    128
#define H_    128
#define D2_   256
#define BS_   (B_*S_)     /* 1536 */
#define NROWS_ (BS_+1)    /* 1537 */

typedef __attribute__((ext_vector_type(16))) _Float16 v16h;
typedef __attribute__((ext_vector_type(8)))  _Float16 v8h;
typedef __attribute__((ext_vector_type(8)))  float    v8f;

__device__ __forceinline__ float sigf(float x) { return 1.0f / (1.0f + __expf(-x)); }
__device__ __forceinline__ float tanh_fast(float x) { return 1.0f - 2.0f / (__expf(2.0f * x) + 1.0f); }

__device__ __forceinline__ v8f wmma_f16(v16h a, v16h b, v8f c) {
    return __builtin_amdgcn_wmma_f32_16x16x32_f16(false, a, false, b, (short)0, c, false, false);
}

// A fragment (16x32 f16) from LDS, row-major [32][256] f16 tile.
// Lane layout (ISA 7.12.2): M = lane&15, element e -> K = 16*(e>>3) + 8*(lane>>4) + (e&7)
__device__ __forceinline__ v16h load_a_frag(const _Float16* As, int stride, int rowbase, int kbase, int lane) {
    int r  = rowbase + (lane & 15);
    int kh = lane >> 4;
    const v8h* p0 = (const v8h*)(As + r * stride + kbase + 8 * kh);
    const v8h* p1 = (const v8h*)(As + r * stride + kbase + 8 * kh + 16);
    v8h lo = *p0, hi = *p1;
    v16h a;
#pragma unroll
    for (int e = 0; e < 8; e++) { a[e] = lo[e]; a[e + 8] = hi[e]; }
    return a;
}

// Pack W (N x K, row-major, f32) into WMMA-B fragments:
// dst[((kt*NT + nt)*512 + lane*16 + e)] = (f16) W[(nt*16 + (lane&15))*K + kt*32 + 16*(e>>3) + 8*(lane>>4) + (e&7)]
__global__ void pack_w_kernel(const float* __restrict__ W, _Float16* __restrict__ dst, int N, int K) {
    int idx = blockIdx.x * 256 + threadIdx.x;
    if (idx >= N * K) return;
    int e = idx & 15;
    int lane = (idx >> 4) & 31;
    int t = idx >> 9;
    int NT = N >> 4;
    int nt = t % NT;
    int kt = t / NT;
    int n = nt * 16 + (lane & 15);
    int k = kt * 32 + 16 * (e >> 3) + 8 * (lane >> 4) + (e & 7);
    dst[idx] = (_Float16)W[(size_t)n * K + k];
}

// Fused LSTM step (both directions): gates = [emb | h_prev](32x256) x Wpacked(256x512) + b
__global__ void lstm_step_kernel(
    const int* __restrict__ enc_batch, const float* __restrict__ emb_table,
    const _Float16* __restrict__ Wf, const _Float16* __restrict__ Wb,
    const float* __restrict__ bf, const float* __restrict__ bb,
    const _Float16* __restrict__ hin_f, _Float16* __restrict__ hout_f, float* __restrict__ c_f,
    const _Float16* __restrict__ hin_b, _Float16* __restrict__ hout_b, float* __restrict__ c_b,
    int t)
{
    __shared__ float Gs[32 * 512];          // 64KB; first 16KB aliased as A-tile
    _Float16* As = (_Float16*)Gs;           // [32][256] f16

    int blk = blockIdx.x;
    int dir = blk / 48;
    int m0  = (blk % 48) * 32;
    int tt  = dir ? (T_ - 1 - t) : t;
    const _Float16* Wp   = dir ? Wb : Wf;
    const float*    bias = dir ? bb : bf;
    const _Float16* hin  = dir ? hin_b : hin_f;
    _Float16*       hout = dir ? hout_b : hout_f;
    float*          cbuf = dir ? c_b : c_f;

    int tid = threadIdx.x;
    {   // stage A: cols 0..127 embedding gather (f32->f16), cols 128..255 h_prev (f16)
        int r  = tid >> 3;
        int c0 = (tid & 7) * 32;
        int n  = m0 + r;
        if (c0 < 128) {
            int tok = enc_batch[n * T_ + tt];
            const float* src = emb_table + (size_t)tok * E_ + c0;
#pragma unroll
            for (int k = 0; k < 32; k++) As[r * 256 + c0 + k] = (_Float16)src[k];
        } else {
            const _Float16* src = hin + (size_t)n * H_ + (c0 - 128);
#pragma unroll
            for (int k = 0; k < 32; k++) As[r * 256 + c0 + k] = src[k];
        }
    }
    __syncthreads();

    int wv = tid >> 5, lane = tid & 31;
    int rt = wv & 1;        // row tile
    int cg = wv >> 1;       // column group of 128 (== one gate block)
    v8f acc[8];
#pragma unroll
    for (int j = 0; j < 8; j++)
#pragma unroll
        for (int e = 0; e < 8; e++) acc[j][e] = 0.0f;

#pragma unroll
    for (int kt = 0; kt < 8; kt++) {
        v16h a = load_a_frag(As, 256, rt * 16, kt * 32, lane);
#pragma unroll
        for (int j = 0; j < 8; j++) {
            int nt = cg * 8 + j;
            v16h b = *(const v16h*)(Wp + ((size_t)(kt * 32 + nt) * 512 + lane * 16));
            acc[j] = wmma_f16(a, b, acc[j]);
        }
    }
    __syncthreads();        // done with As before overwriting Gs
    {   // scatter C tiles (+bias) into Gs
        int lh = lane >> 4, ln = lane & 15;
#pragma unroll
        for (int j = 0; j < 8; j++) {
            int ncol = (cg * 8 + j) * 16 + ln;
            float bv = bias[ncol];
#pragma unroll
            for (int r = 0; r < 8; r++) {
                int row = rt * 16 + r + 8 * lh;
                Gs[row * 512 + ncol] = acc[j][r] + bv;
            }
        }
    }
    __syncthreads();
    {   // gate nonlinearities: c' = sig(f)*c + sig(i)*tanh(g); h' = sig(o)*tanh(c')
        for (int it = 0; it < 16; it++) {
            int e  = tid + it * 256;
            int r  = e >> 7;
            int jh = e & 127;
            int n  = m0 + r;
            float ig = Gs[r * 512 + jh];
            float fg = Gs[r * 512 + 128 + jh];
            float gg = Gs[r * 512 + 256 + jh];
            float og = Gs[r * 512 + 384 + jh];
            float cp = cbuf[(size_t)n * H_ + jh];
            float cn = sigf(fg) * cp + sigf(ig) * tanh_fast(gg);
            float hn = sigf(og) * tanh_fast(cn);
            cbuf[(size_t)n * H_ + jh] = cn;
            hout[(size_t)n * H_ + jh] = (_Float16)hn;
        }
    }
}

// c_red/h_red reduce GEMMs with relu -> hidden rows 1..1536
__global__ void reduce_kernel(
    const float* __restrict__ cf, const float* __restrict__ cb,
    const _Float16* __restrict__ hf, const _Float16* __restrict__ hb,
    const _Float16* __restrict__ Wc_p, const _Float16* __restrict__ Whr_p,
    const float* __restrict__ bc, const float* __restrict__ bhr,
    float* __restrict__ hidden)
{
    __shared__ _Float16 As[32 * 256];
    int blk = blockIdx.x;
    int which = blk / 48;              // 0: c_red (cols 0..127), 1: h_red (cols 128..255)
    int m0 = (blk % 48) * 32;
    int tid = threadIdx.x;
    {
        int r = tid >> 3;
        int c0 = (tid & 7) * 32;
        int n = m0 + r;
        if (which == 0) {
            const float* s = (c0 < 128) ? (cf + (size_t)n * H_ + c0) : (cb + (size_t)n * H_ + (c0 - 128));
#pragma unroll
            for (int k = 0; k < 32; k++) As[r * 256 + c0 + k] = (_Float16)s[k];
        } else {
            const _Float16* s = (c0 < 128) ? (hf + (size_t)n * H_ + c0) : (hb + (size_t)n * H_ + (c0 - 128));
#pragma unroll
            for (int k = 0; k < 32; k++) As[r * 256 + c0 + k] = s[k];
        }
    }
    __syncthreads();
    const _Float16* Wp = which ? Whr_p : Wc_p;
    const float* bias  = which ? bhr : bc;
    int coloff = which ? 128 : 0;
    int wv = tid >> 5, lane = tid & 31;
    int rt  = wv & 1;
    int ntb = (wv >> 1) * 2;
    v8f acc[2];
#pragma unroll
    for (int j = 0; j < 2; j++)
#pragma unroll
        for (int e = 0; e < 8; e++) acc[j][e] = 0.0f;
#pragma unroll
    for (int kt = 0; kt < 8; kt++) {
        v16h a = load_a_frag(As, 256, rt * 16, kt * 32, lane);
#pragma unroll
        for (int j = 0; j < 2; j++) {
            int nt = ntb + j;
            v16h b = *(const v16h*)(Wp + ((size_t)(kt * 8 + nt) * 512 + lane * 16));
            acc[j] = wmma_f16(a, b, acc[j]);
        }
    }
    int lh = lane >> 4, ln = lane & 15;
#pragma unroll
    for (int j = 0; j < 2; j++) {
        int ncol = (ntb + j) * 16 + ln;
        float bv = bias[ncol];
#pragma unroll
        for (int r = 0; r < 8; r++) {
            int row = m0 + rt * 16 + r + 8 * lh;
            float v = acc[j][r] + bv;
            hidden[(size_t)(1 + row) * 256 + coloff + ncol] = fmaxf(v, 0.0f);
        }
    }
}

// XP = hidden @ Wi.T + bi ; HP = hidden @ Wh.T + bh   (1537x256 @ 256x768)
__global__ void proj_kernel(
    const float* __restrict__ hidden,
    const _Float16* __restrict__ Wi_p, const _Float16* __restrict__ Wh_p,
    const float* __restrict__ bi, const float* __restrict__ bh,
    float* __restrict__ XP, float* __restrict__ HP)
{
    __shared__ _Float16 As[32 * 256];
    int m0  = blockIdx.x * 32;
    int cg  = blockIdx.y;   // 0..5 -> 128 output cols each
    int sel = blockIdx.z;   // 0: XP/Wi, 1: HP/Wh
    int tid = threadIdx.x;
    {
        int r = tid >> 3;
        int c0 = (tid & 7) * 32;
        int n = m0 + r;
        if (n < NROWS_) {
            const float* s = hidden + (size_t)n * 256 + c0;
#pragma unroll
            for (int k = 0; k < 32; k++) As[r * 256 + c0 + k] = (_Float16)s[k];
        } else {
#pragma unroll
            for (int k = 0; k < 32; k++) As[r * 256 + c0 + k] = (_Float16)0.0f;
        }
    }
    __syncthreads();
    const _Float16* Wp = sel ? Wh_p : Wi_p;
    const float* bias  = sel ? bh : bi;
    float* out         = sel ? HP : XP;
    int wv = tid >> 5, lane = tid & 31;
    int rt  = wv & 1;
    int ntb = cg * 8 + (wv >> 1) * 2;
    v8f acc[2];
#pragma unroll
    for (int j = 0; j < 2; j++)
#pragma unroll
        for (int e = 0; e < 8; e++) acc[j][e] = 0.0f;
#pragma unroll
    for (int kt = 0; kt < 8; kt++) {
        v16h a = load_a_frag(As, 256, rt * 16, kt * 32, lane);
#pragma unroll
        for (int j = 0; j < 2; j++) {
            int nt = ntb + j;
            v16h b = *(const v16h*)(Wp + ((size_t)(kt * 48 + nt) * 512 + lane * 16));
            acc[j] = wmma_f16(a, b, acc[j]);
        }
    }
    int lh = lane >> 4, ln = lane & 15;
#pragma unroll
    for (int j = 0; j < 2; j++) {
        int ncol = (ntb + j) * 16 + ln;
        float bv = bias[ncol];
#pragma unroll
        for (int r = 0; r < 8; r++) {
            int row = m0 + rt * 16 + r + 8 * lh;
            if (row < NROWS_)
                out[(size_t)row * 768 + ncol] = acc[j][r] + bv;
        }
    }
}

// GRU elementwise on gathered *projected* rows + masked sum over one structure axis.
// dir==0 (cp): out(b,j) = sum_i mask * gru(x=child, h=parent); dir==1 (pc): out(b,i) = sum_j, x=parent,h=child
__global__ void gru_reduce_kernel(
    const int* __restrict__ parent, const int* __restrict__ child,
    const float* __restrict__ mask,
    const float* __restrict__ XP, const float* __restrict__ HP,
    const float* __restrict__ hidden,
    float* __restrict__ tmp, int dir)
{
    int b = blockIdx.x;
    int o = blockIdx.y;
    int d = threadIdx.x;
    float acc = 0.0f;
    for (int s = 0; s < S_; s++) {
        int i = dir ? o : s;
        int j = dir ? s : o;
        int idx3 = (b * S_ + i) * S_ + j;
        float mv = mask[idx3];
        if (mv != 0.0f) {
            int pr = parent[idx3];
            int cr = child[idx3];
            int xr = dir ? pr : cr;
            int hr = dir ? cr : pr;
            const float* xp = XP + (size_t)xr * 768;
            const float* hp = HP + (size_t)hr * 768;
            float ir  = xp[d], iz = xp[256 + d], inn = xp[512 + d];
            float hrr = hp[d], hz = hp[256 + d], hn  = hp[512 + d];
            float h   = hidden[(size_t)hr * 256 + d];
            float r = sigf(ir + hrr);
            float z = sigf(iz + hz);
            float nn = tanh_fast(inn + r * hn);
            acc += mv * ((1.0f - z) * nn + z * h);
        }
    }
    tmp[(size_t)(b * S_ + o) * 256 + d] = acc;
}

// hidden[1+row] += tmp[row] * (||tmp[row]||+0.25)/(||tmp[row]||+1)
__global__ void norm_update_kernel(const float* __restrict__ tmp, float* __restrict__ hidden) {
    __shared__ float red[256];
    int row = blockIdx.x;
    int d = threadIdx.x;
    float v = tmp[(size_t)row * 256 + d];
    red[d] = v * v;
    __syncthreads();
    for (int st = 128; st > 0; st >>= 1) {
        if (d < st) red[d] += red[d + st];
        __syncthreads();
    }
    float nrm = sqrtf(red[0]);
    float scale = (nrm + 0.25f) / (nrm + 1.0f);
    hidden[(size_t)(1 + row) * 256 + d] += v * scale;
}

extern "C" void kernel_launch(void* const* d_in, const int* in_sizes, int n_in,
                              void* d_out, int out_size, void* d_ws, size_t ws_size,
                              hipStream_t stream) {
    (void)in_sizes; (void)n_in; (void)out_size; (void)ws_size;
    const int*   enc_batch = (const int*)d_in[0];
    const int*   parent    = (const int*)d_in[1];
    const int*   child     = (const int*)d_in[2];
    const float* mask      = (const float*)d_in[3];
    const float* emb_table = (const float*)d_in[4];
    const float* Wih_f = (const float*)d_in[5],  *Whh_f = (const float*)d_in[6],  *b_f = (const float*)d_in[7];
    const float* Wih_b = (const float*)d_in[8],  *Whh_b = (const float*)d_in[9],  *b_b = (const float*)d_in[10];
    const float* Wc    = (const float*)d_in[11], *bc    = (const float*)d_in[12];
    const float* Whr   = (const float*)d_in[13], *bhr   = (const float*)d_in[14];
    const float* Wi_cp = (const float*)d_in[15], *Wh_cp = (const float*)d_in[16];
    const float* bi_cp = (const float*)d_in[17], *bh_cp = (const float*)d_in[18];
    const float* Wi_pc = (const float*)d_in[19], *Wh_pc = (const float*)d_in[20];
    const float* bi_pc = (const float*)d_in[21], *bh_pc = (const float*)d_in[22];
    float* hidden = (float*)d_out;

    // --- workspace carve ---
    char* p = (char*)d_ws;
    auto alloc = [&](size_t bytes) -> char* {
        char* r = p;
        p += (bytes + 255) & ~(size_t)255;
        return r;
    };
    _Float16* Wf_p   = (_Float16*)alloc((size_t)8 * 32 * 512 * 2);   // LSTM fwd [Wih|Whh]
    _Float16* Wb_p   = (_Float16*)alloc((size_t)8 * 32 * 512 * 2);
    _Float16* Wc_p   = (_Float16*)alloc((size_t)8 * 8 * 512 * 2);
    _Float16* Whr_p  = (_Float16*)alloc((size_t)8 * 8 * 512 * 2);
    _Float16* Wicp_p = (_Float16*)alloc((size_t)8 * 48 * 512 * 2);
    _Float16* Whcp_p = (_Float16*)alloc((size_t)8 * 48 * 512 * 2);
    _Float16* Wipc_p = (_Float16*)alloc((size_t)8 * 48 * 512 * 2);
    _Float16* Whpc_p = (_Float16*)alloc((size_t)8 * 48 * 512 * 2);
    _Float16* hA_f   = (_Float16*)alloc((size_t)BS_ * H_ * 2);
    _Float16* hB_f   = (_Float16*)alloc((size_t)BS_ * H_ * 2);
    _Float16* hA_b   = (_Float16*)alloc((size_t)BS_ * H_ * 2);
    _Float16* hB_b   = (_Float16*)alloc((size_t)BS_ * H_ * 2);
    float*    c_f    = (float*)alloc((size_t)BS_ * H_ * 4);
    float*    c_b    = (float*)alloc((size_t)BS_ * H_ * 4);
    float*    XP     = (float*)alloc((size_t)NROWS_ * 768 * 4);
    float*    HP     = (float*)alloc((size_t)NROWS_ * 768 * 4);
    float*    tmpb   = (float*)alloc((size_t)BS_ * 256 * 4);

    // --- pack weights to f16 WMMA-B layout (every call; deterministic) ---
    pack_w_kernel<<<256, 256, 0, stream>>>(Wih_f, Wf_p, 512, 128);
    pack_w_kernel<<<256, 256, 0, stream>>>(Whh_f, Wf_p + (size_t)4 * 32 * 512, 512, 128);
    pack_w_kernel<<<256, 256, 0, stream>>>(Wih_b, Wb_p, 512, 128);
    pack_w_kernel<<<256, 256, 0, stream>>>(Whh_b, Wb_p + (size_t)4 * 32 * 512, 512, 128);
    pack_w_kernel<<<128, 256, 0, stream>>>(Wc, Wc_p, 128, 256);
    pack_w_kernel<<<128, 256, 0, stream>>>(Whr, Whr_p, 128, 256);
    pack_w_kernel<<<768, 256, 0, stream>>>(Wi_cp, Wicp_p, 768, 256);
    pack_w_kernel<<<768, 256, 0, stream>>>(Wh_cp, Whcp_p, 768, 256);
    pack_w_kernel<<<768, 256, 0, stream>>>(Wi_pc, Wipc_p, 768, 256);
    pack_w_kernel<<<768, 256, 0, stream>>>(Wh_pc, Whpc_p, 768, 256);

    // --- zero initial LSTM state and hidden row 0 ---
    hipMemsetAsync(hA_f, 0, (size_t)BS_ * H_ * 2, stream);
    hipMemsetAsync(hA_b, 0, (size_t)BS_ * H_ * 2, stream);
    hipMemsetAsync(c_f, 0, (size_t)BS_ * H_ * 4, stream);
    hipMemsetAsync(c_b, 0, (size_t)BS_ * H_ * 4, stream);
    hipMemsetAsync(hidden, 0, 256 * sizeof(float), stream);

    // --- LSTM: 50 fused steps (both directions per launch) ---
    for (int t = 0; t < T_; t++) {
        const _Float16* in_f = (t & 1) ? hB_f : hA_f;
        _Float16*       out_f = (t & 1) ? hA_f : hB_f;
        const _Float16* in_b = (t & 1) ? hB_b : hA_b;
        _Float16*       out_b = (t & 1) ? hA_b : hB_b;
        lstm_step_kernel<<<96, 256, 0, stream>>>(enc_batch, emb_table, Wf_p, Wb_p, b_f, b_b,
                                                 in_f, out_f, c_f, in_b, out_b, c_b, t);
    }
    const _Float16* fin_f = ((T_ - 1) & 1) ? hA_f : hB_f;
    const _Float16* fin_b = ((T_ - 1) & 1) ? hA_b : hB_b;

    // --- reduce to enc_states -> hidden rows 1..1536 ---
    reduce_kernel<<<96, 256, 0, stream>>>(c_f, c_b, fin_f, fin_b, Wc_p, Whr_p, bc, bhr, hidden);

    // --- GRU propagation: project distinct rows, then gather-elementwise-reduce ---
    for (int it = 0; it < 2; it++) {
        proj_kernel<<<dim3(49, 6, 2), 256, 0, stream>>>(hidden, Wicp_p, Whcp_p, bi_cp, bh_cp, XP, HP);
        gru_reduce_kernel<<<dim3(B_, S_), 256, 0, stream>>>(parent, child, mask, XP, HP, hidden, tmpb, 0);
        norm_update_kernel<<<BS_, 256, 0, stream>>>(tmpb, hidden);
    }
    for (int it = 0; it < 2; it++) {
        proj_kernel<<<dim3(49, 6, 2), 256, 0, stream>>>(hidden, Wipc_p, Whpc_p, bi_pc, bh_pc, XP, HP);
        gru_reduce_kernel<<<dim3(B_, S_), 256, 0, stream>>>(parent, child, mask, XP, HP, hidden, tmpb, 1);
        norm_update_kernel<<<BS_, 256, 0, stream>>>(tmpb, hidden);
    }
}
